// PTMHunYuanMoE_62508954026517
// MI455X (gfx1250) — matrix operand order repaired and checked by reading, not compiled, not run on hardware
//
#include <hip/hip_runtime.h>
#include <stdint.h>

// ---- problem constants (from reference setup_inputs) ----
#define B_   2
#define S_   2048      // T*H*W = 8*16*16
#define D_   2048
#define E_   16
#define I_   2048
#define N_   4096      // B*S
#define CAP_ 640       // TOPK * ceil(N/E * 1.25) = 2*320

typedef __attribute__((ext_vector_type(16))) __bf16          v16bf;
typedef __attribute__((ext_vector_type(8)))  float           v8f;
typedef __attribute__((ext_vector_type(8)))  unsigned short  v8us;

union Frag16 { v16bf bf; v8us us[2]; };

__device__ __forceinline__ unsigned short f2bf(float f) {
    unsigned u = __float_as_uint(f);
    return (unsigned short)((u + 0x7FFFu + ((u >> 16) & 1u)) >> 16); // RNE
}

// ------------------------------------------------------------------
// 1) Router: logits -> softmax -> top-2 -> normalized weights
//    one wave per token (8 waves / block)
// ------------------------------------------------------------------
__global__ __launch_bounds__(256)
void hy_router_kernel(const float* __restrict__ x, const float* __restrict__ wg,
                      int* __restrict__ idx, float* __restrict__ wt) {
    const int wave = threadIdx.x >> 5;
    const int lane = threadIdx.x & 31;
    const int n = blockIdx.x * 8 + wave;
    const float* xp = x + ((size_t)((n & 1) * S_ + (n >> 1))) * D_;

    float acc[E_];
#pragma unroll
    for (int e = 0; e < E_; ++e) acc[e] = 0.f;

    for (int d = lane; d < D_; d += 32) {
        const float xv = xp[d];
#pragma unroll
        for (int e = 0; e < E_; ++e)
            acc[e] = fmaf(xv, wg[e * D_ + d], acc[e]);
    }
#pragma unroll
    for (int e = 0; e < E_; ++e) {
#pragma unroll
        for (int m = 16; m >= 1; m >>= 1)
            acc[e] += __shfl_xor(acc[e], m, 32);
    }

    if (lane == 0) {
        float mx = acc[0];
#pragma unroll
        for (int e = 1; e < E_; ++e) mx = fmaxf(mx, acc[e]);
        float g[E_], s = 0.f;
#pragma unroll
        for (int e = 0; e < E_; ++e) { g[e] = __expf(acc[e] - mx); s += g[e]; }
        const float inv = 1.f / s;
#pragma unroll
        for (int e = 0; e < E_; ++e) g[e] *= inv;

        int e0 = 0; float g0 = g[0];
#pragma unroll
        for (int e = 1; e < E_; ++e) if (g[e] > g0) { g0 = g[e]; e0 = e; }
        int e1 = -1; float g1 = -1.f;
#pragma unroll
        for (int e = 0; e < E_; ++e) if (e != e0 && g[e] > g1) { g1 = g[e]; e1 = e; }

        const float denom = fmaxf(g0 + g1, 1.1920929e-07f);
        idx[n * 2 + 0] = e0;  idx[n * 2 + 1] = e1;
        wt [n * 2 + 0] = g0 / denom;  wt[n * 2 + 1] = g1 / denom;
    }
}

// ------------------------------------------------------------------
// 2) Slot assignment: reproduce the reference's k-major cumsum
//    priority. Single block, LDS histogram + prefix.
// ------------------------------------------------------------------
__global__ __launch_bounds__(256)
void hy_assign_kernel(const int* __restrict__ idx,
                      int* __restrict__ slot, int* __restrict__ src) {
    __shared__ int hist[2][256][E_];   // 32 KB
    const int t = threadIdx.x;
    const int per = N_ / 256;          // 16 tokens per thread

    for (int i = t; i < E_ * CAP_; i += 256) src[i] = -1;
    __threadfence_block();

#pragma unroll
    for (int k = 0; k < 2; ++k)
        for (int e = 0; e < E_; ++e) hist[k][t][e] = 0;

#pragma unroll
    for (int k = 0; k < 2; ++k)
        for (int j = 0; j < per; ++j) {
            const int n = t * per + j;
            hist[k][t][idx[n * 2 + k]] += 1;
        }
    __syncthreads();

    if (t < E_) {
        const int e = t;
        int run = 0;
        for (int i = 0; i < 256; ++i) { int v = hist[0][i][e]; hist[0][i][e] = run; run += v; }
        // k=1 offsets continue after ALL k=0 entries (k-major cumsum)
        for (int i = 0; i < 256; ++i) { int v = hist[1][i][e]; hist[1][i][e] = run; run += v; }
    }
    __syncthreads();

#pragma unroll
    for (int k = 0; k < 2; ++k)
        for (int j = 0; j < per; ++j) {
            const int n = t * per + j;
            const int e = idx[n * 2 + k];
            const int p = hist[k][t][e]++;
            slot[n * 2 + k] = (p < CAP_) ? p : -1;
            if (p < CAP_) src[e * CAP_ + p] = n;
        }
}

// ------------------------------------------------------------------
// 3) Dispatch: gather tokens into bf16 Xb (E*CAP, D); zero empty slots
// ------------------------------------------------------------------
__global__ __launch_bounds__(256)
void hy_dispatch_kernel(const float* __restrict__ x, const int* __restrict__ src,
                        unsigned short* __restrict__ Xb) {
    const int ec = blockIdx.x;              // e*CAP + c
    const int n = src[ec];
    const int d0 = threadIdx.x * 8;
    v8us o;
    if (n >= 0) {
        const float* xp = x + ((size_t)((n & 1) * S_ + (n >> 1))) * D_ + d0;
        const float4 a = *(const float4*)(xp);
        const float4 b = *(const float4*)(xp + 4);
        o[0] = f2bf(a.x); o[1] = f2bf(a.y); o[2] = f2bf(a.z); o[3] = f2bf(a.w);
        o[4] = f2bf(b.x); o[5] = f2bf(b.y); o[6] = f2bf(b.z); o[7] = f2bf(b.w);
    } else {
#pragma unroll
        for (int i = 0; i < 8; ++i) o[i] = 0;
    }
    *(v8us*)(Xb + (size_t)ec * D_ + d0) = o;
}

// ------------------------------------------------------------------
// 4/5) Tiled WMMA GEMM.  out[c, o] = sum_k A[c,k] * W[o,k]
//  A: bf16 (E*CAP, K) row-major.   W: fp32 (E, Nout, K) row-major,
//  converted to bf16 while staging into LDS.
//  FUSED: two weight tensors (gate, up); epilogue silu(g)*u -> bf16.
//  else : single weight tensor; epilogue fp32 store.
//  Block = 256 thr = 8 waves; block tile 64(c) x 64(o); K-step 32.
//  Wave w: rows 16*(w&3), cols 32*(w>>2) (two 16x16 col tiles).
// ------------------------------------------------------------------
template <bool FUSED>
__global__ __launch_bounds__(256)
void hy_moe_gemm_kernel(const unsigned short* __restrict__ A,
                        const float* __restrict__ W0,
                        const float* __restrict__ W1,
                        unsigned short* __restrict__ OutBf,
                        float* __restrict__ OutF,
                        int K, int Nout) {
    constexpr int KT  = 32;
    constexpr int LDR = KT + 8;   // 40 halves: 16B-aligned rows, conflict padding
    __shared__ unsigned short sA [64 * LDR];
    __shared__ unsigned short sB0[64 * LDR];
    __shared__ unsigned short sB1[FUSED ? 64 * LDR : 8];

    const int e     = blockIdx.z;
    const int crow0 = blockIdx.y * 64;
    const int ncol0 = blockIdx.x * 64;
    const int tid   = threadIdx.x;
    const int lane  = tid & 31;
    const int wave  = tid >> 5;
    const int wrow  = (wave & 3) * 16;
    const int wcol  = (wave >> 2) * 32;

    const unsigned short* Ap  = A  + ((size_t)(e * CAP_ + crow0)) * K;
    const float*          W0p = W0 + ((size_t)e * Nout + ncol0) * K;
    const float*          W1p = FUSED ? (W1 + ((size_t)e * Nout + ncol0) * K) : W0;

    v8f acc0[2] = {};
    v8f acc1[2] = {};

    const int row = tid >> 2;        // 0..63 staging row
    const int ch  = (tid & 3) * 8;   // 0,8,16,24 (halves / floats)

    const int m16   = lane & 15;
    const int kkA   = (lane < 16) ? 0 : 16;
    const int kkB   = (lane < 16) ? 0 : 16;

    for (int k0 = 0; k0 < K; k0 += KT) {
        // ---- stage A (already bf16): one 16B load/store per thread ----
        *(v8us*)(sA + row * LDR + ch) = *(const v8us*)(Ap + (size_t)row * K + k0 + ch);
        // ---- stage W0 (+W1): fp32 -> bf16 ----
        {
            const float* p = W0p + (size_t)row * K + k0 + ch;
            const float4 a = *(const float4*)p;
            const float4 b = *(const float4*)(p + 4);
            v8us o;
            o[0] = f2bf(a.x); o[1] = f2bf(a.y); o[2] = f2bf(a.z); o[3] = f2bf(a.w);
            o[4] = f2bf(b.x); o[5] = f2bf(b.y); o[6] = f2bf(b.z); o[7] = f2bf(b.w);
            *(v8us*)(sB0 + row * LDR + ch) = o;
        }
        if (FUSED) {
            const float* p = W1p + (size_t)row * K + k0 + ch;
            const float4 a = *(const float4*)p;
            const float4 b = *(const float4*)(p + 4);
            v8us o;
            o[0] = f2bf(a.x); o[1] = f2bf(a.y); o[2] = f2bf(a.z); o[3] = f2bf(a.w);
            o[4] = f2bf(b.x); o[5] = f2bf(b.y); o[6] = f2bf(b.z); o[7] = f2bf(b.w);
            *(v8us*)(sB1 + row * LDR + ch) = o;
        }
        // prefetch next K tile while this one is consumed
        if (k0 + KT < K) {
            __builtin_prefetch(Ap  + (size_t)row * K + k0 + KT + ch, 0, 3);
            __builtin_prefetch(W0p + (size_t)row * K + k0 + KT + ch, 0, 3);
            if (FUSED) __builtin_prefetch(W1p + (size_t)row * K + k0 + KT + ch, 0, 3);
        }
        __syncthreads();

        // ---- A fragment (16x32 bf16): lane<16 holds K 0-15 ----
        Frag16 af;
        {
            const unsigned short* p = sA + (wrow + m16) * LDR + kkA;
            af.us[0] = *(const v8us*)(p);
            af.us[1] = *(const v8us*)(p + 8);
        }
#pragma unroll
        for (int ct = 0; ct < 2; ++ct) {
            // ---- B fragment (32x16): lane holds 16 contiguous K of col N ----
            Frag16 b0;
            {
                const unsigned short* p = sB0 + (wcol + ct * 16 + m16) * LDR + kkB;
                b0.us[0] = *(const v8us*)(p);
                b0.us[1] = *(const v8us*)(p + 8);
            }
            acc0[ct] = __builtin_amdgcn_wmma_f32_16x16x32_bf16(
                false, af.bf, false, b0.bf, (short)0, acc0[ct], false, false);
            if (FUSED) {
                Frag16 b1;
                const unsigned short* p = sB1 + (wcol + ct * 16 + m16) * LDR + kkB;
                b1.us[0] = *(const v8us*)(p);
                b1.us[1] = *(const v8us*)(p + 8);
                acc1[ct] = __builtin_amdgcn_wmma_f32_16x16x32_bf16(
                    false, af.bf, false, b1.bf, (short)0, acc1[ct], false, false);
            }
        }
        __syncthreads();
    }

    // ---- epilogue: C/D layout = VGPR r: lanes<16 -> M=r, lanes>=16 -> M=r+8
    const int mbase = (lane < 16) ? 0 : 8;
    const int nn    = lane & 15;
#pragma unroll
    for (int ct = 0; ct < 2; ++ct) {
#pragma unroll
        for (int r = 0; r < 8; ++r) {
            const size_t orow = (size_t)(e * CAP_ + crow0 + wrow + r + mbase);
            const size_t ocol = (size_t)(ncol0 + wcol + ct * 16 + nn);
            if (FUSED) {
                const float g = acc0[ct][r];
                const float u = acc1[ct][r];
                const float h = (g / (1.f + __expf(-g))) * u;   // silu(g)*u
                OutBf[orow * Nout + ocol] = f2bf(h);
            } else {
                OutF[orow * Nout + ocol] = acc0[ct][r];
            }
        }
    }
}

// ------------------------------------------------------------------
// 6) Combine: out[n] = w0*Yd[e0,s0] + w1*Yd[e1,s1]
// ------------------------------------------------------------------
__global__ __launch_bounds__(256)
void hy_combine_kernel(const float* __restrict__ Yd, const int* __restrict__ idx,
                       const int* __restrict__ slot, const float* __restrict__ wt,
                       float* __restrict__ out) {
    const int n  = blockIdx.x;
    const int d0 = threadIdx.x * 8;
    const int e0 = idx[n * 2 + 0], e1 = idx[n * 2 + 1];
    const int s0 = slot[n * 2 + 0], s1 = slot[n * 2 + 1];
    const float w0 = wt[n * 2 + 0], w1 = wt[n * 2 + 1];

    float r[8];
#pragma unroll
    for (int i = 0; i < 8; ++i) r[i] = 0.f;

    if (s0 >= 0) {
        const float* p = Yd + ((size_t)(e0 * CAP_ + s0)) * D_ + d0;
        const float4 a = *(const float4*)p;
        const float4 b = *(const float4*)(p + 4);
        r[0] += w0 * a.x; r[1] += w0 * a.y; r[2] += w0 * a.z; r[3] += w0 * a.w;
        r[4] += w0 * b.x; r[5] += w0 * b.y; r[6] += w0 * b.z; r[7] += w0 * b.w;
    }
    if (s1 >= 0) {
        const float* p = Yd + ((size_t)(e1 * CAP_ + s1)) * D_ + d0;
        const float4 a = *(const float4*)p;
        const float4 b = *(const float4*)(p + 4);
        r[0] += w1 * a.x; r[1] += w1 * a.y; r[2] += w1 * a.z; r[3] += w1 * a.w;
        r[4] += w1 * b.x; r[5] += w1 * b.y; r[6] += w1 * b.z; r[7] += w1 * b.w;
    }
    float* op = out + ((size_t)((n & 1) * S_ + (n >> 1))) * D_ + d0;
    float4 A = { r[0], r[1], r[2], r[3] };
    float4 Bv = { r[4], r[5], r[6], r[7] };
    *(float4*)(op)     = A;
    *(float4*)(op + 4) = Bv;
}

// ------------------------------------------------------------------
extern "C" void kernel_launch(void* const* d_in, const int* in_sizes, int n_in,
                              void* d_out, int out_size, void* d_ws, size_t ws_size,
                              hipStream_t stream) {
    const float* x  = (const float*)d_in[0];   // (B,T,H,W,D) fp32
    const float* wg = (const float*)d_in[1];   // (E,D)
    const float* gp = (const float*)d_in[2];   // (E,I,D)
    const float* up = (const float*)d_in[3];   // (E,I,D)
    const float* dp = (const float*)d_in[4];   // (E,D,I)
    float* out = (float*)d_out;

    char* ws = (char*)d_ws;
    size_t off = 0;
    auto alloc = [&](size_t bytes) -> void* {
        void* p = ws + off;
        off += bytes;
        off = (off + 255) & ~(size_t)255;
        return p;
    };
    int*            idx  = (int*)            alloc((size_t)N_ * 2 * sizeof(int));
    float*          wt   = (float*)          alloc((size_t)N_ * 2 * sizeof(float));
    int*            slot = (int*)            alloc((size_t)N_ * 2 * sizeof(int));
    int*            src  = (int*)            alloc((size_t)E_ * CAP_ * sizeof(int));
    unsigned short* Xb   = (unsigned short*) alloc((size_t)E_ * CAP_ * D_ * 2);
    unsigned short* Hb   = (unsigned short*) alloc((size_t)E_ * CAP_ * I_ * 2);
    float*          Yd   = (float*)          alloc((size_t)E_ * CAP_ * D_ * 4);
    (void)ws_size; (void)in_sizes; (void)n_in; (void)out_size;

    hy_router_kernel<<<N_ / 8, 256, 0, stream>>>(x, wg, idx, wt);
    hy_assign_kernel<<<1, 256, 0, stream>>>(idx, slot, src);
    hy_dispatch_kernel<<<E_ * CAP_, 256, 0, stream>>>(x, src, Xb);

    dim3 g1(I_ / 64, CAP_ / 64, E_);
    hy_moe_gemm_kernel<true><<<g1, 256, 0, stream>>>(Xb, gp, up, Hb, nullptr, D_, I_);

    dim3 g2(D_ / 64, CAP_ / 64, E_);
    hy_moe_gemm_kernel<false><<<g2, 256, 0, stream>>>(Hb, dp, nullptr, nullptr, Yd, I_, D_);

    hy_combine_kernel<<<N_, 256, 0, stream>>>(Yd, idx, slot, wt, out);
}
